// DeepSeekV3Model_57939108823119
// MI455X (gfx1250) — compile-verified, matrix operand order checked by reading, not run on hardware
//
#include <hip/hip_runtime.h>

#define EXPERTS 8
#define TOPK 2

typedef __bf16 bf16;
typedef bf16  v16bf __attribute__((ext_vector_type(16)));
typedef float v8f   __attribute__((ext_vector_type(8)));

struct alignas(16) U16B { unsigned u[4]; };
struct alignas(16) F4   { float x, y, z, w; };
union  FragB { v16bf v; U16B q[2]; };
union  Pack8 { bf16 h[8]; U16B q; };

__device__ __forceinline__ bf16 f2bf(float f) {
  union { float f; unsigned u; } c; c.f = f;
  unsigned r = c.u + 0x7FFFu + ((c.u >> 16) & 1u);   // round-to-nearest-even
  unsigned short hs = (unsigned short)(r >> 16);
  return __builtin_bit_cast(bf16, hs);
}

__device__ __forceinline__ void cvt8(Pack8& pk, F4 a, F4 b) {
  pk.h[0] = f2bf(a.x); pk.h[1] = f2bf(a.y); pk.h[2] = f2bf(a.z); pk.h[3] = f2bf(a.w);
  pk.h[4] = f2bf(b.x); pk.h[5] = f2bf(b.y); pk.h[6] = f2bf(b.z); pk.h[7] = f2bf(b.w);
}

// Async 16-byte global -> LDS copy (CDNA5 TDM-adjacent path, ASYNCcnt-tracked).
__device__ __forceinline__ void async_copy16(bf16* lds, const bf16* g) {
  unsigned l = (unsigned)(size_t)lds;                 // LDS byte offset (low 32b)
  unsigned long long ga = (unsigned long long)(size_t)g;
  asm volatile("global_load_async_to_lds_b128 %0, %1, off"
               :: "v"(l), "v"(ga) : "memory");
}
__device__ __forceinline__ void wait_async() {
  asm volatile("s_wait_asynccnt 0x0" ::: "memory");
}

// -------------------------------------------------------------- convert ----
__global__ __launch_bounds__(256)
void convert_f32_bf16_kernel(const float* __restrict__ src, bf16* __restrict__ dst,
                             long long n) {
  long long i = ((long long)blockIdx.x * 256 + threadIdx.x) * 8;
  if (i + 8 <= n) {
    F4 a = *(const F4*)(src + i);
    F4 b = *(const F4*)(src + i + 4);
    Pack8 pk; cvt8(pk, a, b);
    *(U16B*)(dst + i) = pk.q;
  } else {
    for (; i < n; ++i) dst[i] = f2bf(src[i]);
  }
}

// ---------------------------------------------------------------- router ----
__global__ __launch_bounds__(256)
void router_kernel(const float* __restrict__ x, const float* __restrict__ gw,
                   float* __restrict__ comb, int T, int D) {
  __shared__ float gwS[EXPERTS * 1024];
  const int tid = threadIdx.x;
  for (int i = tid; i < EXPERTS * D; i += 256) gwS[i] = gw[i];
  __syncthreads();
  const int lane = tid & 31, wid = tid >> 5;
  for (int j = 0; j < 8; ++j) {                 // 64 tokens / block, 8 waves
    const int t = blockIdx.x * 64 + wid * 8 + j;
    const float* xr = x + (size_t)t * D;
    float acc[EXPERTS];
#pragma unroll
    for (int e = 0; e < EXPERTS; ++e) acc[e] = 0.f;
    for (int d0 = lane * 4; d0 < D; d0 += 128) {
      F4 xv = *(const F4*)(xr + d0);
      float xs[4] = {xv.x, xv.y, xv.z, xv.w};
#pragma unroll
      for (int c = 0; c < 4; ++c)
#pragma unroll
        for (int e = 0; e < EXPERTS; ++e)
          acc[e] += xs[c] * gwS[e * D + d0 + c];
    }
#pragma unroll
    for (int e = 0; e < EXPERTS; ++e)
      for (int off = 16; off >= 1; off >>= 1)
        acc[e] += __shfl_xor(acc[e], off, 32);
    if (lane == 0) {
      float mx = acc[0];
#pragma unroll
      for (int e = 1; e < EXPERTS; ++e) mx = fmaxf(mx, acc[e]);
      float p[EXPERTS], s = 0.f;
#pragma unroll
      for (int e = 0; e < EXPERTS; ++e) { p[e] = __expf(acc[e] - mx); s += p[e]; }
      const float inv = 1.f / s;
#pragma unroll
      for (int e = 0; e < EXPERTS; ++e) p[e] *= inv;
      int i1 = 0;
#pragma unroll
      for (int e = 1; e < EXPERTS; ++e) if (p[e] > p[i1]) i1 = e;
      int i2 = (i1 == 0) ? 1 : 0;
#pragma unroll
      for (int e = 0; e < EXPERTS; ++e) if (e != i1 && p[e] > p[i2]) i2 = e;
      const float wsum = p[i1] + p[i2];
#pragma unroll
      for (int e = 0; e < EXPERTS; ++e) comb[(size_t)t * EXPERTS + e] = 0.f;
      comb[(size_t)t * EXPERTS + i1] = p[i1] / wsum;
      comb[(size_t)t * EXPERTS + i2] = p[i2] / wsum;
    }
  }
}

// --------------------------------------------------------- list building ----
__global__ void build_lists_kernel(const float* __restrict__ comb,
                                   int* __restrict__ toks, float* __restrict__ wgts,
                                   int* __restrict__ counts, int T) {
  const int e = blockIdx.x, lane = threadIdx.x;
  int count = 0;
  for (int base = 0; base < T; base += 32) {
    const int t = base + lane;
    const float w = comb[(size_t)t * EXPERTS + e];
    const bool pr = w > 0.f;
    const unsigned mask = (unsigned)__ballot(pr);
    const int pre = __popc(mask & ((1u << lane) - 1u));
    if (pr) {
      toks[(size_t)e * T + count + pre] = t;
      wgts[(size_t)e * T + count + pre] = w;
    }
    count += __popc(mask);
  }
  if (lane == 0) counts[e] = count;
}

__global__ void scan_kernel(const int* __restrict__ counts, int* __restrict__ offs) {
  int s = 0;
  for (int e = 0; e < EXPERTS; ++e) { offs[e] = s; s += counts[e]; }
}

// ------------------------------------------------------------- gate + up ----
// 128x64x32 tiles, double-buffered LDS, async G->LDS staging, 8 WMMA/K-step.
__global__ __launch_bounds__(256)
void gateup_kernel(const bf16* __restrict__ XB,
                   const float* __restrict__ WgF, const float* __restrict__ WuF,
                   const bf16* __restrict__ WgB, const bf16* __restrict__ WuB,
                   const int* __restrict__ toks, const int* __restrict__ counts,
                   const int* __restrict__ offsets, bf16* __restrict__ act,
                   int T, int D, int H) {
  const int e = blockIdx.z;
  const int count = counts ? counts[e] : T;
  const int tile_m = blockIdx.y;
  if (tile_m * 128 >= count) return;
  const int row_off = offsets ? offsets[e] : 0;
  const int* tlist = toks ? toks + (size_t)e * T : nullptr;
  const int hbase = blockIdx.x * 64;
  const size_t wbase = (size_t)e * H * D;

  __shared__ bf16 As[2][128][40];
  __shared__ bf16 Bgs[2][64][40];
  __shared__ bf16 Bus[2][64][40];

  const int tid = threadIdx.x;
  const int lane = tid & 31, wid = tid >> 5;
  const int wm = wid & 3, wn = wid >> 2;      // 4 m-strips(32) x 2 n-strips(32)
  const int lm = lane & 15, kg = lane >> 4;

  // A staging: thread copies 16 bf16 of one of 128 rows
  const int ra = tid >> 1, ca = (tid & 1) * 16;
  int pa_idx = tile_m * 128 + ra;
  if (pa_idx >= count) pa_idx = count - 1;
  const int t_ld = tlist ? tlist[pa_idx] : pa_idx;
  const bf16* xrow = XB + (size_t)t_ld * D;

  // B staging: thread handles 8 elems of one of 64 rows
  const int rb = tid >> 2, cb = (tid & 3) * 8;
  const float* growF = WgF + wbase + (size_t)(hbase + rb) * D;
  const float* urowF = WuF + wbase + (size_t)(hbase + rb) * D;
  const bf16*  growB = WgB ? WgB + wbase + (size_t)(hbase + rb) * D : nullptr;
  const bf16*  urowB = WuB ? WuB + wbase + (size_t)(hbase + rb) * D : nullptr;

  const v8f vz = {0.f, 0.f, 0.f, 0.f, 0.f, 0.f, 0.f, 0.f};
  v8f accG[2][2] = {{vz, vz}, {vz, vz}}, accU[2][2] = {{vz, vz}, {vz, vz}};

  auto stage = [&](int buf, int kk) {
    if (growB) {                              // pure bf16: async G->LDS path
      async_copy16(&As[buf][ra][ca],     xrow + kk + ca);
      async_copy16(&As[buf][ra][ca + 8], xrow + kk + ca + 8);
      async_copy16(&Bgs[buf][rb][cb],    growB + kk + cb);
      async_copy16(&Bus[buf][rb][cb],    urowB + kk + cb);
    } else {                                  // fp32 fallback: convert in VGPRs
      *(U16B*)&As[buf][ra][ca]     = *(const U16B*)(xrow + kk + ca);
      *(U16B*)&As[buf][ra][ca + 8] = *(const U16B*)(xrow + kk + ca + 8);
      Pack8 pk;
      cvt8(pk, *(const F4*)(growF + kk + cb), *(const F4*)(growF + kk + cb + 4));
      *(U16B*)&Bgs[buf][rb][cb] = pk.q;
      cvt8(pk, *(const F4*)(urowF + kk + cb), *(const F4*)(urowF + kk + cb + 4));
      *(U16B*)&Bus[buf][rb][cb] = pk.q;
    }
    if (kk + 64 < D) {                        // warm WGP$/L2 two K-steps ahead
      __builtin_prefetch(xrow + kk + 64 + ca, 0, 3);
      if (growB) {
        __builtin_prefetch(growB + kk + 64 + cb, 0, 3);
        __builtin_prefetch(urowB + kk + 64 + cb, 0, 3);
      } else {
        __builtin_prefetch(growF + kk + 64 + cb, 0, 3);
        __builtin_prefetch(urowF + kk + 64 + cb, 0, 3);
      }
    }
  };

  stage(0, 0);
  wait_async();
  __syncthreads();
  const int nk = D / 32;
  for (int ki = 0; ki < nk; ++ki) {
    const int buf = ki & 1;
    if (ki + 1 < nk) stage(buf ^ 1, (ki + 1) * 32);
    FragB fa[2];
#pragma unroll
    for (int mi = 0; mi < 2; ++mi) {
      const bf16* pa = &As[buf][wm * 32 + mi * 16 + lm][kg * 8];
      fa[mi].q[0] = *(const U16B*)pa;
      fa[mi].q[1] = *(const U16B*)(pa + 16);
    }
#pragma unroll
    for (int ni = 0; ni < 2; ++ni) {
      FragB fg, fu;
      const bf16* pg = &Bgs[buf][wn * 32 + ni * 16 + lm][kg * 8];
      fg.q[0] = *(const U16B*)pg; fg.q[1] = *(const U16B*)(pg + 16);
      const bf16* pu = &Bus[buf][wn * 32 + ni * 16 + lm][kg * 8];
      fu.q[0] = *(const U16B*)pu; fu.q[1] = *(const U16B*)(pu + 16);
#pragma unroll
      for (int mi = 0; mi < 2; ++mi) {
        accG[mi][ni] = __builtin_amdgcn_wmma_f32_16x16x32_bf16(
            false, fa[mi].v, false, fg.v, (short)0, accG[mi][ni], false, false);
        accU[mi][ni] = __builtin_amdgcn_wmma_f32_16x16x32_bf16(
            false, fa[mi].v, false, fu.v, (short)0, accU[mi][ni], false, false);
      }
    }
    wait_async();                             // own async stage ops done
    __syncthreads();
  }
  // fused SiLU(g)*u epilogue -> bf16 activation rows
#pragma unroll
  for (int mi = 0; mi < 2; ++mi)
#pragma unroll
    for (int ni = 0; ni < 2; ++ni)
#pragma unroll
      for (int i = 0; i < 8; ++i) {
        const int p = tile_m * 128 + wm * 32 + mi * 16 + i + 8 * kg;
        if (p < count) {
          const float g = accG[mi][ni][i], u = accU[mi][ni][i];
          const float s = g * __builtin_amdgcn_rcpf(1.f + __expf(-g));
          const int hcol = hbase + wn * 32 + ni * 16 + lm;
          act[(size_t)(row_off + p) * H + hcol] = f2bf(s * u);
        }
      }
}

// ------------------------------------------------------------------ down ----
__global__ __launch_bounds__(256)
void down_kernel(const bf16* __restrict__ act,
                 const float* __restrict__ WdF, const bf16* __restrict__ WdB,
                 const int* __restrict__ toks, const float* __restrict__ wgts,
                 const int* __restrict__ counts, const int* __restrict__ offsets,
                 float* __restrict__ out, int T, int D, int H, int use_atomic) {
  const int e = blockIdx.z;
  const int count = counts ? counts[e] : T;
  const int tile_m = blockIdx.y;
  if (tile_m * 128 >= count) return;
  const int row_off = offsets ? offsets[e] : 0;
  const int* tlist = toks ? toks + (size_t)e * T : nullptr;
  const float* wlist = wgts ? wgts + (size_t)e * T : nullptr;
  const int dbase = blockIdx.x * 64;
  const size_t wbase = (size_t)e * D * H;

  __shared__ bf16 Aa[2][128][40];
  __shared__ bf16 Bw[2][64][40];

  const int tid = threadIdx.x;
  const int lane = tid & 31, wid = tid >> 5;
  const int wm = wid & 3, wn = wid >> 2;
  const int lm = lane & 15, kg = lane >> 4;

  const int ra = tid >> 1, ca = (tid & 1) * 16;
  int pa_idx = tile_m * 128 + ra;
  if (pa_idx >= count) pa_idx = count - 1;
  const bf16* arow = act + (size_t)(row_off + pa_idx) * H;

  const int rb = tid >> 2, cb = (tid & 3) * 8;
  const float* wrowF = WdF + wbase + (size_t)(dbase + rb) * H;
  const bf16*  wrowB = WdB ? WdB + wbase + (size_t)(dbase + rb) * H : nullptr;

  const v8f vz = {0.f, 0.f, 0.f, 0.f, 0.f, 0.f, 0.f, 0.f};
  v8f acc[2][2] = {{vz, vz}, {vz, vz}};

  auto stage = [&](int buf, int kk) {
    if (wrowB) {
      async_copy16(&Aa[buf][ra][ca],     arow + kk + ca);
      async_copy16(&Aa[buf][ra][ca + 8], arow + kk + ca + 8);
      async_copy16(&Bw[buf][rb][cb],     wrowB + kk + cb);
    } else {
      *(U16B*)&Aa[buf][ra][ca]     = *(const U16B*)(arow + kk + ca);
      *(U16B*)&Aa[buf][ra][ca + 8] = *(const U16B*)(arow + kk + ca + 8);
      Pack8 pk;
      cvt8(pk, *(const F4*)(wrowF + kk + cb), *(const F4*)(wrowF + kk + cb + 4));
      *(U16B*)&Bw[buf][rb][cb] = pk.q;
    }
    if (kk + 64 < H) {
      __builtin_prefetch(arow + kk + 64 + ca, 0, 3);
      if (wrowB) __builtin_prefetch(wrowB + kk + 64 + cb, 0, 3);
      else       __builtin_prefetch(wrowF + kk + 64 + cb, 0, 3);
    }
  };

  stage(0, 0);
  wait_async();
  __syncthreads();
  const int nk = H / 32;
  for (int ki = 0; ki < nk; ++ki) {
    const int buf = ki & 1;
    if (ki + 1 < nk) stage(buf ^ 1, (ki + 1) * 32);
    FragB fa[2];
#pragma unroll
    for (int mi = 0; mi < 2; ++mi) {
      const bf16* pa = &Aa[buf][wm * 32 + mi * 16 + lm][kg * 8];
      fa[mi].q[0] = *(const U16B*)pa;
      fa[mi].q[1] = *(const U16B*)(pa + 16);
    }
#pragma unroll
    for (int ni = 0; ni < 2; ++ni) {
      FragB fb;
      const bf16* pb = &Bw[buf][wn * 32 + ni * 16 + lm][kg * 8];
      fb.q[0] = *(const U16B*)pb; fb.q[1] = *(const U16B*)(pb + 16);
#pragma unroll
      for (int mi = 0; mi < 2; ++mi)
        acc[mi][ni] = __builtin_amdgcn_wmma_f32_16x16x32_bf16(
            false, fa[mi].v, false, fb.v, (short)0, acc[mi][ni], false, false);
    }
    wait_async();
    __syncthreads();
  }
#pragma unroll
  for (int mi = 0; mi < 2; ++mi)
#pragma unroll
    for (int i = 0; i < 8; ++i) {
      const int p = tile_m * 128 + wm * 32 + mi * 16 + i + 8 * kg;
      if (p < count) {
        const int t = tlist ? tlist[p] : p;
        const float w = wlist ? wlist[p] : 1.f;
#pragma unroll
        for (int ni = 0; ni < 2; ++ni) {
          const int col = dbase + wn * 32 + ni * 16 + lm;
          const float v = acc[mi][ni][i] * w;
          float* dst = out + (size_t)t * D + col;
          if (use_atomic) atomicAdd(dst, v);
          else            *dst = v;
        }
      }
    }
}

// ---------------------------------------------------------------- launch ----
extern "C" void kernel_launch(void* const* d_in, const int* in_sizes, int n_in,
                              void* d_out, int out_size, void* d_ws, size_t ws_size,
                              hipStream_t stream) {
  (void)n_in; (void)out_size;
  const float* x      = (const float*)d_in[0];
  const float* gate_w = (const float*)d_in[1];
  const float* shg    = (const float*)d_in[2];
  const float* shu    = (const float*)d_in[3];
  const float* shd    = (const float*)d_in[4];
  const float* eg     = (const float*)d_in[5];
  const float* eu     = (const float*)d_in[6];
  const float* ed     = (const float*)d_in[7];
  float* out = (float*)d_out;

  const int D = in_sizes[1] / EXPERTS;  // gate_w: [E, D]
  const int T = in_sizes[0] / D;        // x: [T, D]
  const int H = in_sizes[2] / D;        // sh_gate: [H, D]
  const size_t nSh = (size_t)H * D;     // one shared-weight tensor
  const size_t nEx = (size_t)EXPERTS * H * D;

  auto pad = [](size_t b) { return (b + 255) & ~(size_t)255; };
  const size_t need_base =
      pad((size_t)T * EXPERTS * 4) + 2 * pad((size_t)EXPERTS * T * 4) + 2 * pad(256) +
      pad((size_t)T * H * 2) + pad((size_t)TOPK * T * H * 2) + pad((size_t)T * D * 2);
  const size_t need_w = 3 * pad(nSh * 2) + 3 * pad(nEx * 2);
  const bool wconv = (ws_size >= need_base + need_w);  // bf16 weight cache fits?

  char* w = (char*)d_ws;
  auto take = [&](size_t b) { char* p = w; w += (b + 255) & ~(size_t)255; return p; };
  float* comb   = (float*)take((size_t)T * EXPERTS * 4);
  int*   toks   = (int*)  take((size_t)EXPERTS * T * 4);
  float* wgts   = (float*)take((size_t)EXPERTS * T * 4);
  int*   counts = (int*)  take(256);
  int*   offs   = (int*)  take(256);
  bf16*  actS   = (bf16*) take((size_t)T * H * 2);
  bf16*  actR   = (bf16*) take((size_t)TOPK * T * H * 2);
  bf16*  xb     = (bf16*) take((size_t)T * D * 2);
  bf16 *shgB = nullptr, *shuB = nullptr, *shdB = nullptr;
  bf16 *egB = nullptr, *euB = nullptr, *edB = nullptr;
  if (wconv) {
    shgB = (bf16*)take(nSh * 2); shuB = (bf16*)take(nSh * 2); shdB = (bf16*)take(nSh * 2);
    egB  = (bf16*)take(nEx * 2); euB  = (bf16*)take(nEx * 2); edB  = (bf16*)take(nEx * 2);
  }

  auto cvt = [&](const float* s, bf16* d, size_t n) {
    convert_f32_bf16_kernel<<<dim3((unsigned)((n / 8 + 255) / 256)), 256, 0, stream>>>(
        s, d, (long long)n);
  };
  cvt(x, xb, (size_t)T * D);
  if (wconv) {
    cvt(shg, shgB, nSh); cvt(shu, shuB, nSh); cvt(shd, shdB, nSh);
    cvt(eg, egB, nEx);   cvt(eu, euB, nEx);   cvt(ed, edB, nEx);
  }

  router_kernel<<<dim3(T / 64), 256, 0, stream>>>(x, gate_w, comb, T, D);
  build_lists_kernel<<<dim3(EXPERTS), 32, 0, stream>>>(comb, toks, wgts, counts, T);
  scan_kernel<<<1, 1, 0, stream>>>(counts, offs);

  const int mt = (T + 127) / 128;
  const dim3 blk(256);
  gateup_kernel<<<dim3(H / 64, mt, 1), blk, 0, stream>>>(
      xb, shg, shu, shgB, shuB, nullptr, nullptr, nullptr, actS, T, D, H);
  gateup_kernel<<<dim3(H / 64, mt, EXPERTS), blk, 0, stream>>>(
      xb, eg, eu, egB, euB, toks, counts, offs, actR, T, D, H);
  // shared-down stores (initializes out), routed-down scatter-adds after it.
  down_kernel<<<dim3(D / 64, mt, 1), blk, 0, stream>>>(
      actS, shd, shdB, nullptr, nullptr, nullptr, nullptr, out, T, D, H, 0);
  down_kernel<<<dim3(D / 64, mt, EXPERTS), blk, 0, stream>>>(
      actR, ed, edB, toks, wgts, counts, offs, out, T, D, H, 1);
}
